// MultiHeadAttention_74131135529884
// MI455X (gfx1250) — matrix operand order repaired and checked by reading, not compiled, not run on hardware
//
#include <hip/hip_runtime.h>

#define DEV __device__ __forceinline__

typedef __attribute__((ext_vector_type(16))) _Float16 v16h;
typedef __attribute__((ext_vector_type(8)))  _Float16 h8;
typedef __attribute__((ext_vector_type(8)))  float    v8f;
typedef __attribute__((ext_vector_type(4)))  float    f32x4;

static constexpr int Bb = 2, S = 2048, D = 1024, H = 16, HD = 64;
static constexpr float SCALE = 0.125f; // 1/sqrt(64)

// ---- WMMA fragment loaders (CDNA5 16-bit layouts, wave32) ----
// A (16x32 f16): lane L (0-15) row=L holds K 0..7 & 16..23; lane L+16 holds K 8..15 & 24..31.
DEV v16h ldA(const _Float16* tile, int stride, int row_off, int lane) {
  const _Float16* p = tile + (row_off + (lane & 15)) * stride + ((lane >> 4) * 8);
  h8 lo = *(const h8*)(p);
  h8 hi = *(const h8*)(p + 16);
  v16h a;
#pragma unroll
  for (int i = 0; i < 8; ++i) { a[i] = lo[i]; a[i + 8] = hi[i]; }
  return a;
}
// B (32x16 f16) from N-major (transposed) LDS tile: lane n holds K=(lane>>4)*8 + 0..7 & 16+(lane>>4)*8 + 0..7, col n=lane&15.
DEV v16h ldB(const _Float16* tileT, int stride, int col_off, int k_off, int lane) {
  const _Float16* p = tileT + (col_off + (lane & 15)) * stride + k_off + ((lane >> 4) * 8);
  h8 lo = *(const h8*)(p);
  h8 hi = *(const h8*)(p + 16);
  v16h b;
#pragma unroll
  for (int i = 0; i < 8; ++i) { b[i] = lo[i]; b[i + 8] = hi[i]; }
  return b;
}

DEV v8f wmma16(v16h a, v16h b, v8f c) {
  return __builtin_amdgcn_wmma_f32_16x16x32_f16(false, a, false, b, (short)0, c, false, false);
}

DEV float rmax16(float v) {
#pragma unroll
  for (int m = 1; m < 16; m <<= 1) v = fmaxf(v, __shfl_xor(v, m, 32));
  return v;
}
DEV float rsum16(float v) {
#pragma unroll
  for (int m = 1; m < 16; m <<= 1) v += __shfl_xor(v, m, 32);
  return v;
}

// ---------------------------------------------------------------------------
// GEMM: Out[M=4096,N=1024] = X[4096,1024] @ W[1024,1024] + bias
// Workgroup tile 128x64, 8 waves (4x2), 32x32 per wave = 2x2 WMMA tiles.
// ---------------------------------------------------------------------------
template <bool IN_HALF, bool OUT_F32>
__global__ __launch_bounds__(256) void proj_gemm(const void* __restrict__ Xv,
                                                 const float* __restrict__ W,
                                                 const float* __restrict__ bias,
                                                 void* __restrict__ Outv) {
  constexpr int ASTR = 40, BSTR = 40;
  __shared__ __align__(16) _Float16 As[128 * ASTR];
  __shared__ __align__(16) _Float16 Bt[64 * BSTR];
  const int tid = threadIdx.x;
  const int lane = tid & 31, w = tid >> 5;
  const int wm = w & 3, wn = w >> 2;
  const int m0 = blockIdx.x * 128, n0 = blockIdx.y * 64;

  v8f acc[2][2] = {};
  for (int kk = 0; kk < 1024; kk += 32) {
    if constexpr (IN_HALF) {
      const _Float16* X16 = (const _Float16*)Xv;
#pragma unroll
      for (int i = 0; i < 2; ++i) {
        int slot = tid + i * 256;
        int row = slot >> 2, grp = slot & 3;
        h8 v = *(const h8*)(X16 + (size_t)(m0 + row) * 1024 + kk + grp * 8);
        *(h8*)(&As[row * ASTR + grp * 8]) = v;
      }
    } else {
      const float* X = (const float*)Xv;
#pragma unroll
      for (int i = 0; i < 4; ++i) {
        int slot = tid + i * 256;
        int row = slot >> 3, grp = slot & 7;
        f32x4 v = *(const f32x4*)(X + (size_t)(m0 + row) * 1024 + kk + grp * 4);
#pragma unroll
        for (int c = 0; c < 4; ++c) As[row * ASTR + grp * 4 + c] = (_Float16)v[c];
      }
    }
    // W tile [32 k x 64 n] fp32 -> f16, stored N-major (transposed) for B-frags
#pragma unroll
    for (int i = 0; i < 2; ++i) {
      int slot = tid + i * 256;
      int k = slot >> 4, ng = slot & 15;
      f32x4 v = *(const f32x4*)(W + (size_t)(kk + k) * 1024 + n0 + ng * 4);
#pragma unroll
      for (int c = 0; c < 4; ++c) Bt[(ng * 4 + c) * BSTR + k] = (_Float16)v[c];
    }
    __syncthreads();
#pragma unroll
    for (int tj = 0; tj < 2; ++tj) {
      v16h b = ldB(Bt, BSTR, wn * 32 + tj * 16, 0, lane);
#pragma unroll
      for (int ti = 0; ti < 2; ++ti) {
        v16h a = ldA(As, ASTR, wm * 32 + ti * 16, lane);
        acc[ti][tj] = wmma16(a, b, acc[ti][tj]);
      }
    }
    __syncthreads();
  }
#pragma unroll
  for (int ti = 0; ti < 2; ++ti)
#pragma unroll
    for (int tj = 0; tj < 2; ++tj) {
      int n = n0 + wn * 32 + tj * 16 + (lane & 15);
      float bv = bias[n];
#pragma unroll
      for (int j = 0; j < 8; ++j) {
        int m = m0 + wm * 32 + ti * 16 + ((lane >> 4) * 8) + j;
        float val = acc[ti][tj][j] + bv;
        if constexpr (OUT_F32) ((float*)Outv)[(size_t)m * 1024 + n] = val;
        else                   ((_Float16*)Outv)[(size_t)m * 1024 + n] = (_Float16)val;
      }
    }
}

// ---------------------------------------------------------------------------
// Fused attention: one workgroup = 32 query rows of one (b,h).
// Pass A: QK^T tiles -> online (rowmax, rowsum). Pass B: recompute scores
// (K from L2), write normalized weights once, P(f16) @ V via WMMA.
// ---------------------------------------------------------------------------
__global__ __launch_bounds__(256) void attn_fused(const _Float16* __restrict__ Q16,
                                                  const _Float16* __restrict__ K16,
                                                  const _Float16* __restrict__ V16,
                                                  float* __restrict__ attn,
                                                  _Float16* __restrict__ C16) {
  constexpr int QSTR = 72, KSTR = 72, VSTR = 136, PSTR = 136;
  __shared__ __align__(16) _Float16 Qs[32 * QSTR];
  __shared__ __align__(16) _Float16 Ks[128 * KSTR];
  __shared__ __align__(16) _Float16 Vst[64 * VSTR];   // transposed: [d][k]
  __shared__ __align__(16) _Float16 Ps[32 * PSTR];
  __shared__ float redM[8][32], redL[8][32];
  __shared__ float gM[32], gInvL[32];

  const int tid = threadIdx.x, lane = tid & 31, w = tid >> 5;
  const int q0 = blockIdx.x * 32, hy = blockIdx.y, bz = blockIdx.z;
  const size_t rowbase = (size_t)bz * 2048;
  const size_t hcol = (size_t)hy * 64;

  { // stage Q block [32 x 64] once
    int row = tid >> 3, dg = tid & 7;
    h8 v = *(const h8*)(Q16 + (rowbase + q0 + row) * 1024 + hcol + dg * 8);
    *(h8*)(&Qs[row * QSTR + dg * 8]) = v;
  }
  __syncthreads();

  float rm[2][8], rl[2][8];
#pragma unroll
  for (int qt = 0; qt < 2; ++qt)
#pragma unroll
    for (int j = 0; j < 8; ++j) { rm[qt][j] = -3.0e38f; rl[qt][j] = 0.f; }

  // ---- Pass A ----
  for (int ch = 0; ch < 16; ++ch) {
#pragma unroll
    for (int i = 0; i < 4; ++i) {
      int slot = tid + i * 256;
      int kr = slot >> 3, dg = slot & 7;
      h8 v = *(const h8*)(K16 + (rowbase + ch * 128 + kr) * 1024 + hcol + dg * 8);
      *(h8*)(&Ks[kr * KSTR + dg * 8]) = v;
    }
    if (ch < 15)
      __builtin_prefetch(K16 + (rowbase + (ch + 1) * 128 + (tid >> 1)) * 1024 + hcol + (tid & 1) * 32, 0, 1);
    __syncthreads();
    v8f cs[2] = {};
#pragma unroll
    for (int ks = 0; ks < 2; ++ks) {
      v16h b = ldB(Ks, KSTR, w * 16, ks * 32, lane);
#pragma unroll
      for (int qt = 0; qt < 2; ++qt) {
        v16h a = ldA(Qs + ks * 32, QSTR, qt * 16, lane);
        cs[qt] = wmma16(a, b, cs[qt]);
      }
    }
#pragma unroll
    for (int qt = 0; qt < 2; ++qt)
#pragma unroll
      for (int j = 0; j < 8; ++j) {
        float s = cs[qt][j] * SCALE;
        float tm = rmax16(s);
        float nm = fmaxf(rm[qt][j], tm);
        float e = __expf(s - nm);
        float ts = rsum16(e);
        rl[qt][j] = rl[qt][j] * __expf(rm[qt][j] - nm) + ts;
        rm[qt][j] = nm;
      }
    __syncthreads();
  }

  // cross-wave (m, l) combine
  if ((lane & 15) == 0) {
    int g = lane >> 4;
#pragma unroll
    for (int qt = 0; qt < 2; ++qt)
#pragma unroll
      for (int j = 0; j < 8; ++j) {
        redM[w][qt * 16 + g * 8 + j] = rm[qt][j];
        redL[w][qt * 16 + g * 8 + j] = rl[qt][j];
      }
  }
  __syncthreads();
  if (tid < 32) {
    float m = -3.0e38f;
#pragma unroll
    for (int ww = 0; ww < 8; ++ww) m = fmaxf(m, redM[ww][tid]);
    float L = 0.f;
#pragma unroll
    for (int ww = 0; ww < 8; ++ww) L += redL[ww][tid] * __expf(redM[ww][tid] - m);
    gM[tid] = m;
    gInvL[tid] = 1.0f / L;
  }
  __syncthreads();
  float myM[2][8], myI[2][8];
#pragma unroll
  for (int qt = 0; qt < 2; ++qt)
#pragma unroll
    for (int j = 0; j < 8; ++j) {
      int r = qt * 16 + (lane >> 4) * 8 + j;
      myM[qt][j] = gM[r];
      myI[qt][j] = gInvL[r];
    }

  // ---- Pass B ----
  v8f cacc = {};
  const int mi = w >> 2, ni = w & 3;
  float* attn_base = attn + ((size_t)(bz * 16 + hy) * 2048 + q0) * 2048;

  for (int ch = 0; ch < 16; ++ch) {
    __syncthreads();
#pragma unroll
    for (int i = 0; i < 4; ++i) {
      int slot = tid + i * 256;
      int kr = slot >> 3, dg = slot & 7;
      h8 kv = *(const h8*)(K16 + (rowbase + ch * 128 + kr) * 1024 + hcol + dg * 8);
      *(h8*)(&Ks[kr * KSTR + dg * 8]) = kv;
      h8 vv = *(const h8*)(V16 + (rowbase + ch * 128 + kr) * 1024 + hcol + dg * 8);
#pragma unroll
      for (int c = 0; c < 8; ++c) Vst[(dg * 8 + c) * VSTR + kr] = vv[c];
    }
    __syncthreads();
    v8f cs[2] = {};
#pragma unroll
    for (int ks = 0; ks < 2; ++ks) {
      v16h b = ldB(Ks, KSTR, w * 16, ks * 32, lane);
#pragma unroll
      for (int qt = 0; qt < 2; ++qt) {
        v16h a = ldA(Qs + ks * 32, QSTR, qt * 16, lane);
        cs[qt] = wmma16(a, b, cs[qt]);
      }
    }
#pragma unroll
    for (int qt = 0; qt < 2; ++qt)
#pragma unroll
      for (int j = 0; j < 8; ++j) {
        float p = __expf(cs[qt][j] * SCALE - myM[qt][j]) * myI[qt][j];
        int mrow = qt * 16 + (lane >> 4) * 8 + j;
        int kcol = w * 16 + (lane & 15);
        attn_base[(size_t)mrow * 2048 + ch * 128 + kcol] = p;
        Ps[mrow * PSTR + kcol] = (_Float16)p;
      }
    __syncthreads();
#pragma unroll
    for (int ks = 0; ks < 4; ++ks) {
      v16h a = ldA(Ps + ks * 32, PSTR, mi * 16, lane);
      v16h b = ldB(Vst, VSTR, ni * 16, ks * 32, lane);
      cacc = wmma16(a, b, cacc);
    }
  }
#pragma unroll
  for (int j = 0; j < 8; ++j) {
    int mrow = mi * 16 + (lane >> 4) * 8 + j;
    int col = ni * 16 + (lane & 15);
    C16[(rowbase + q0 + mrow) * 1024 + hcol + col] = (_Float16)cacc[j];
  }
}

extern "C" void kernel_launch(void* const* d_in, const int* in_sizes, int n_in,
                              void* d_out, int out_size, void* d_ws, size_t ws_size,
                              hipStream_t stream) {
  (void)in_sizes; (void)n_in; (void)out_size; (void)ws_size;
  const float* q  = (const float*)d_in[0];
  const float* k  = (const float*)d_in[1];
  const float* v  = (const float*)d_in[2];
  const float* Wq = (const float*)d_in[3];
  const float* bq = (const float*)d_in[4];
  const float* Wk = (const float*)d_in[5];
  const float* bk = (const float*)d_in[6];
  const float* Wv = (const float*)d_in[7];
  const float* bv = (const float*)d_in[8];
  const float* Wo = (const float*)d_in[9];
  const float* bo = (const float*)d_in[10];

  char* ws = (char*)d_ws;
  _Float16* Q16 = (_Float16*)(ws);
  _Float16* K16 = (_Float16*)(ws + (size_t)8 * 1024 * 1024);
  _Float16* V16 = (_Float16*)(ws + (size_t)16 * 1024 * 1024);
  _Float16* C16 = (_Float16*)(ws + (size_t)24 * 1024 * 1024);

  float* out  = (float*)d_out;
  float* attn = out + (size_t)Bb * S * D;

  dim3 gproj(32, 16, 1);
  proj_gemm<false, false><<<gproj, 256, 0, stream>>>(q, Wq, bq, Q16);
  proj_gemm<false, false><<<gproj, 256, 0, stream>>>(k, Wk, bk, K16);
  proj_gemm<false, false><<<gproj, 256, 0, stream>>>(v, Wv, bv, V16);
  attn_fused<<<dim3(S / 32, H, Bb), 256, 0, stream>>>(Q16, K16, V16, attn, C16);
  proj_gemm<true, true><<<gproj, 256, 0, stream>>>(C16, Wo, bo, out);
}
